// SNNReadoutModule_20538533609756
// MI455X (gfx1250) — matrix operand order repaired and checked by reading, not compiled, not run on hardware
//
#include <hip/hip_runtime.h>

// ---------------------------------------------------------------------------
// Shapes (fixed by the reference): x[B,T,H] @ W[H,O] -> h2[B*T,O], then a
// sequential EWMA+softmax scan over t producing out[B,O].
// ---------------------------------------------------------------------------
#define SNN_B 64
#define SNN_T 2048
#define SNN_H 1024
#define SNN_O 128
#define SNN_M (SNN_B * SNN_T)     // 131072 GEMM rows

#define ALPHA_F 0.9048374180359595f   // exp(-0.1)

typedef __attribute__((ext_vector_type(16))) _Float16 v16h;
typedef __attribute__((ext_vector_type(8)))  _Float16 v8h;
typedef __attribute__((ext_vector_type(2)))  _Float16 v2h;
typedef __attribute__((ext_vector_type(8)))  float    v8f;

// ---------------------------------------------------------------------------
// Kernel 1: h2 = X * W  via v_wmma_f32_16x16x32_f16.
//   X: [M=131072, K=1024] fp32 row-major, W: [K=1024, N=128] fp32 row-major.
//   Grid: M/128 = 1024 workgroups of 256 threads (8 waves).
//   Each wave: one 16-row M stripe x 8 N-tiles of 16 (full N=128).
// ---------------------------------------------------------------------------
__global__ __launch_bounds__(256) void snn_gemm_wmma(
    const float* __restrict__ X, const float* __restrict__ W,
    float* __restrict__ H2) {
  constexpr int K = SNN_H, N = SNN_O;
  constexpr int LDA = 48;   // padded pitch in halves (96 B, 16B-aligned)
  constexpr int LDB = 48;

  __shared__ __align__(16) _Float16 As[128 * LDA];   // As[m][k]
  __shared__ __align__(16) _Float16 Bs[128 * LDB];   // Bs[n][k]  (W transposed)

  const int tid  = threadIdx.x;
  const int wave = tid >> 5;       // 0..7
  const int lane = tid & 31;
  const int m0   = blockIdx.x * 128;

  v8f acc[8] = {};  // 8 N-tiles of 16x16 f32

  for (int k0 = 0; k0 < K; k0 += 32) {
    __syncthreads();   // previous fragments fully consumed

    // ---- stage A tile: 128 rows x 32 k.  2 threads per row; f32->f16 packed
    //      into two v8h and stored as two ds_store_b128.
    {
      const int r = tid >> 1;
      const int c = (tid & 1) * 16;
      const float* src = X + (size_t)(m0 + r) * K + k0 + c;
      float4 v0 = *(const float4*)(src + 0);
      float4 v1 = *(const float4*)(src + 4);
      float4 v2 = *(const float4*)(src + 8);
      float4 v3 = *(const float4*)(src + 12);
      v8h p0, p1;
      p0[0] = (_Float16)v0.x; p0[1] = (_Float16)v0.y;
      p0[2] = (_Float16)v0.z; p0[3] = (_Float16)v0.w;
      p0[4] = (_Float16)v1.x; p0[5] = (_Float16)v1.y;
      p0[6] = (_Float16)v1.z; p0[7] = (_Float16)v1.w;
      p1[0] = (_Float16)v2.x; p1[1] = (_Float16)v2.y;
      p1[2] = (_Float16)v2.z; p1[3] = (_Float16)v2.w;
      p1[4] = (_Float16)v3.x; p1[5] = (_Float16)v3.y;
      p1[6] = (_Float16)v3.z; p1[7] = (_Float16)v3.w;
      *(v8h*)&As[r * LDA + c]     = p0;
      *(v8h*)&As[r * LDA + c + 8] = p1;
    }
    // ---- stage B tile transposed: Bs[n][k] from W[k][n].  Each thread owns
    //      one k-pair (2kp, 2kp+1) x 8 n values -> 8 packed ds_store_b32.
    //      kp = tid&15 keeps transposed writes on distinct LDS banks; the
    //      strided W reads are gathers served from L2 (W fully resident).
    {
      const int kp = tid & 15;          // k = 2*kp, 2*kp+1
      const int n0 = (tid >> 4) * 8;    // 16 groups * 8 n = 128
      const float* s0 = W + (size_t)(k0 + 2 * kp) * N + n0;
      const float* s1 = s0 + N;
      float4 a0 = *(const float4*)(s0);
      float4 a1 = *(const float4*)(s0 + 4);
      float4 b0 = *(const float4*)(s1);
      float4 b1 = *(const float4*)(s1 + 4);
      const float ev[8] = {a0.x, a0.y, a0.z, a0.w, a1.x, a1.y, a1.z, a1.w};
      const float od[8] = {b0.x, b0.y, b0.z, b0.w, b1.x, b1.y, b1.z, b1.w};
#pragma unroll
      for (int i = 0; i < 8; ++i) {
        v2h p;
        p[0] = (_Float16)ev[i];
        p[1] = (_Float16)od[i];
        *(v2h*)&Bs[(n0 + i) * LDB + 2 * kp] = p;   // dword-aligned
      }
    }
    __syncthreads();

    // ---- A fragment (ISA 7.12.2, 16-bit A 16x32): lane L holds row L&15,
    //      K chunks [8h,8h+8) and [16+8h,16+8h+8), h = L>>4.
    const int r = wave * 16 + (lane & 15);
    const int h = lane >> 4;
    v8h a_lo = *(const v8h*)&As[r * LDA + 8 * h];
    v8h a_hi = *(const v8h*)&As[r * LDA + 16 + 8 * h];
    v16h a;
#pragma unroll
    for (int i = 0; i < 8; ++i) { a[i] = a_lo[i]; a[8 + i] = a_hi[i]; }

    // ---- 8 independent WMMAs (good ILP; distinct accumulators).
#pragma unroll
    for (int j = 0; j < 8; ++j) {
      const int n = j * 16 + (lane & 15);
      v8h b_lo = *(const v8h*)&Bs[n * LDB + 8 * h];
      v8h b_hi = *(const v8h*)&Bs[n * LDB + 16 + 8 * h];
      v16h b;
#pragma unroll
      for (int i = 0; i < 8; ++i) { b[i] = b_lo[i]; b[8 + i] = b_hi[i]; }
      acc[j] = __builtin_amdgcn_wmma_f32_16x16x32_f16(
          false, a, false, b, (short)0, acc[j], false, false);
    }
  }

  // ---- store D: lane L, VGPR v -> row v + 8*(L>>4), col L&15 (per N-tile).
  const int nlo   = lane & 15;
  const int mbase = m0 + wave * 16 + (lane >> 4) * 8;
#pragma unroll
  for (int j = 0; j < 8; ++j) {
    const int n = j * 16 + nlo;
#pragma unroll
    for (int v = 0; v < 8; ++v)
      H2[(size_t)(mbase + v) * N + n] = acc[j][v];
  }
}

// ---------------------------------------------------------------------------
// Kernel 2: sequential scan.  One wave32 per batch row b; each lane owns 4 of
// the O=128 channels in registers.  Softmax reductions via wave shuffles.
// ---------------------------------------------------------------------------
__global__ __launch_bounds__(32) void snn_scan(
    const float* __restrict__ H2, float* __restrict__ Out) {
  constexpr int T = SNN_T, O = SNN_O;
  const int b    = blockIdx.x;
  const int lane = threadIdx.x;
  const float alpha = ALPHA_F;
  const float beta  = 1.0f - ALPHA_F;

  float mem0 = 0.f, mem1 = 0.f, mem2 = 0.f, mem3 = 0.f;
  float out0 = 0.f, out1 = 0.f, out2 = 0.f, out3 = 0.f;

  const float* base = H2 + (size_t)b * T * O + lane * 4;
  float4 h = *(const float4*)base;               // t = 0

  for (int t = 0; t < T - 1; ++t) {
    // register-prefetch next row (row T-1 exists even though unused)
    float4 hn = *(const float4*)(base + (size_t)(t + 1) * O);

    mem0 = alpha * mem0 + beta * h.x;
    mem1 = alpha * mem1 + beta * h.y;
    mem2 = alpha * mem2 + beta * h.z;
    mem3 = alpha * mem3 + beta * h.w;

    // row max over 128 channels: 4 local + 5-step wave32 xor-reduction
    float mx = fmaxf(fmaxf(mem0, mem1), fmaxf(mem2, mem3));
#pragma unroll
    for (int s = 16; s > 0; s >>= 1) mx = fmaxf(mx, __shfl_xor(mx, s, 32));

    float e0 = __expf(mem0 - mx), e1 = __expf(mem1 - mx);
    float e2 = __expf(mem2 - mx), e3 = __expf(mem3 - mx);
    float sum = e0 + e1 + e2 + e3;
#pragma unroll
    for (int s = 16; s > 0; s >>= 1) sum += __shfl_xor(sum, s, 32);

    const float inv = __frcp_rn(sum);
    out0 += e0 * inv; out1 += e1 * inv; out2 += e2 * inv; out3 += e3 * inv;

    h = hn;
  }

  float4 r; r.x = out0; r.y = out1; r.z = out2; r.w = out3;
  *(float4*)(Out + (size_t)b * O + lane * 4) = r;
}

// ---------------------------------------------------------------------------
extern "C" void kernel_launch(void* const* d_in, const int* in_sizes, int n_in,
                              void* d_out, int out_size, void* d_ws,
                              size_t ws_size, hipStream_t stream) {
  (void)in_sizes; (void)n_in; (void)out_size; (void)ws_size;
  const float* x = (const float*)d_in[0];        // [B,T,H] fp32
  const float* w = (const float*)d_in[1];        // [H,O]   fp32
  float* out = (float*)d_out;                    // [B,O]   fp32
  float* h2  = (float*)d_ws;                     // [B*T,O] fp32 = 64 MiB

  snn_gemm_wmma<<<SNN_M / 128, 256, 0, stream>>>(x, w, h2);
  snn_scan<<<SNN_B, 32, 0, stream>>>(h2, out);
}
